// InformerUni_40278203302252
// MI455X (gfx1250) — matrix operand order repaired and checked by reading, not compiled, not run on hardware
//
#include <hip/hip_runtime.h>
#include <math.h>

// ---------------- model dimensions ----------------
#define BATCH    8
#define DMODEL   256
#define NHEADS   8
#define DHEAD    32
#define DFF      512
#define SEQLEN   512
#define LABELLEN 256
#define PREDLEN  256
#define SPACEDIM 512
#define NMARK    4
#define CONVDFF  32

typedef _Float16 f16;
typedef __attribute__((ext_vector_type(16))) _Float16 v16h;
typedef __attribute__((ext_vector_type(8)))  _Float16 v8h;
typedef __attribute__((ext_vector_type(8)))  float    v8f;

// ---------------- WMMA GEMM tile config ----------------
// All GEMMs in this model have K % 4 == 0 and N % 4 == 0 (K,N in
// {16,32,64,128,256,512,768}), so vector-granularity bounds checks are exact.
#define BM 64
#define BN 64
#define BK 64
#define LDSS 72   // f16 elements per LDS row: 144B rows, keeps 16B alignment for v8h loads

__device__ __forceinline__ float geluf(float x) { return 0.5f * x * (1.0f + erff(x * 0.70710678118654752f)); }
__device__ __forceinline__ float eluf(float x)  { return x > 0.f ? x : (expf(x) - 1.f); }

struct GemmP {
    const float* A;     // [batch][M][K] (rows may be circularly shifted)
    const float* B;     // BT==0: [K][N];  BT==1: [N][K]
    const float* bias;  // [N] or null
    const float* R;     // residual, same layout as C, or null
    float* C;
    int M, N, K;
    int lda, ldb, ldc;
    long sA, sB, sC, sR;    // per-batch strides (elements)
    int rowShift, rowMod;   // circular A-row addressing when CIRC
    int act;                // 0 none, 1 gelu, 2 elu
    int accum;              // 1: C += result
};

template<int BT, bool CIRC>
__global__ __launch_bounds__(256) void gemm_f16_wmma(GemmP p)
{
    __shared__ __align__(16) f16 shA[BM * LDSS];
    __shared__ __align__(16) f16 shB[BN * LDSS];   // stored [n][k]

    const int tid  = threadIdx.x;
    const int lane = tid & 31;
    const int wv   = tid >> 5;
    const int wm   = wv & 3;   // m offset 16*wm
    const int wn   = wv >> 2;  // n offset 32*wn
    const int bm0  = blockIdx.y * BM;
    const int bn0  = blockIdx.x * BN;
    const long z   = blockIdx.z;

    const float* A  = p.A + z * p.sA;
    const float* Bg = p.B + z * p.sB;
    float*       C  = p.C + z * p.sC;
    const float* R  = p.R ? (p.R + z * p.sR) : nullptr;

    v8f acc[2];
    for (int i = 0; i < 8; ++i) { acc[0][i] = 0.f; acc[1][i] = 0.f; }

    for (int k0 = 0; k0 < p.K; k0 += BK) {
        // ---- stage A tile (BM x BK) fp32 -> f16, 4 float4 per thread ----
        #pragma unroll
        for (int it = 0; it < 4; ++it) {
            int e = tid + 256 * it;        // float4 unit 0..1023
            int r = e >> 4;                // 0..63
            int c = (e & 15) << 2;         // 0..60
            int gm = bm0 + r;
            int gk = k0 + c;
            float4 v = make_float4(0.f, 0.f, 0.f, 0.f);
            if (gm < p.M && gk < p.K) {    // K%4==0 -> full vector in bounds
                int row = gm;
                if (CIRC) { row += p.rowShift; row %= p.rowMod; if (row < 0) row += p.rowMod; }
                v = *(const float4*)(A + (long)row * p.lda + gk);
            }
            f16* d = &shA[r * LDSS + c];
            d[0] = (f16)v.x; d[1] = (f16)v.y; d[2] = (f16)v.z; d[3] = (f16)v.w;
        }
        // ---- stage B tile as [n][k] ----
        if (BT == 0) {
            // B is [K][N]: coalesced float4 along n, transposed scatter into LDS
            #pragma unroll
            for (int it = 0; it < 4; ++it) {
                int e  = tid + 256 * it;
                int kk = e >> 4;               // 0..63
                int nn = (e & 15) << 2;        // 0..60
                int gk = k0 + kk, gn = bn0 + nn;
                float4 v = make_float4(0.f, 0.f, 0.f, 0.f);
                if (gk < p.K && gn < p.N)      // N%4==0 -> full vector in bounds
                    v = *(const float4*)(Bg + (long)gk * p.ldb + gn);
                shB[(nn + 0) * LDSS + kk] = (f16)v.x;
                shB[(nn + 1) * LDSS + kk] = (f16)v.y;
                shB[(nn + 2) * LDSS + kk] = (f16)v.z;
                shB[(nn + 3) * LDSS + kk] = (f16)v.w;
            }
        } else {
            // B is [N][K]: direct float4 copy
            #pragma unroll
            for (int it = 0; it < 4; ++it) {
                int e = tid + 256 * it;
                int r = e >> 4;
                int c = (e & 15) << 2;
                int gn = bn0 + r;
                int gk = k0 + c;
                float4 v = make_float4(0.f, 0.f, 0.f, 0.f);
                if (gn < p.N && gk < p.K)
                    v = *(const float4*)(Bg + (long)gn * p.ldb + gk);
                f16* d = &shB[r * LDSS + c];
                d[0] = (f16)v.x; d[1] = (f16)v.y; d[2] = (f16)v.z; d[3] = (f16)v.w;
            }
        }
        __syncthreads();

        // ---- fragments per ISA 7.12.2 layouts: 4 WMMAs per stage ----
        {
            const int am = lane & 15;   // A row within subtile / B column
            const int kg = lane >> 4;   // K group selector
            #pragma unroll
            for (int kk = 0; kk < BK; kk += 32) {
                const f16* ar = &shA[(16 * wm + am) * LDSS + kk];
                union { v16h v; v8h h[2]; } af, bf0, bf1;
                af.h[0] = *(const v8h*)&ar[kg * 8];
                af.h[1] = *(const v8h*)&ar[16 + kg * 8];
                const f16* b0 = &shB[(32 * wn + am) * LDSS + kk + kg * 8];
                const f16* b1 = &shB[(32 * wn + 16 + am) * LDSS + kk + kg * 8];
                bf0.h[0] = *(const v8h*)&b0[0]; bf0.h[1] = *(const v8h*)&b0[16];
                bf1.h[0] = *(const v8h*)&b1[0]; bf1.h[1] = *(const v8h*)&b1[16];
                acc[0] = __builtin_amdgcn_wmma_f32_16x16x32_f16(false, af.v, false, bf0.v, (short)0, acc[0], false, false);
                acc[1] = __builtin_amdgcn_wmma_f32_16x16x32_f16(false, af.v, false, bf1.v, (short)0, acc[1], false, false);
            }
        }
        __syncthreads();
    }

    // ---- epilogue: bias + residual + activation + store ----
    const int nl = lane & 15;
    const int mb = (lane >> 4) * 8;
    #pragma unroll
    for (int t = 0; t < 2; ++t) {
        int n = bn0 + 32 * wn + 16 * t + nl;
        if (n >= p.N) continue;
        float bv = p.bias ? p.bias[n] : 0.f;
        #pragma unroll
        for (int r = 0; r < 8; ++r) {
            int m = bm0 + 16 * wm + mb + r;
            if (m >= p.M) continue;
            float x = acc[t][r] + bv;
            if (R) x += R[(long)m * p.ldc + n];
            if (p.act == 1) x = geluf(x);
            else if (p.act == 2) x = eluf(x);
            long off = (long)m * p.ldc + n;
            if (p.accum) C[off] += x; else C[off] = x;
        }
    }
}

// ---------------- elementwise / reduction kernels ----------------
__global__ __launch_bounds__(256) void layernorm_k(const float* x, const float* g, const float* bb, float* y)
{
    __shared__ float s[DMODEL];
    __shared__ float stat[2];
    long row = blockIdx.x;
    int t = threadIdx.x;
    float v = x[row * DMODEL + t];
    s[t] = v; __syncthreads();
    for (int o = 128; o > 0; o >>= 1) { if (t < o) s[t] += s[t + o]; __syncthreads(); }
    if (t == 0) stat[0] = s[0] / DMODEL;
    __syncthreads();
    float m = stat[0];
    float dv = v - m;
    s[t] = dv * dv; __syncthreads();
    for (int o = 128; o > 0; o >>= 1) { if (t < o) s[t] += s[t + o]; __syncthreads(); }
    if (t == 0) stat[1] = s[0] / DMODEL;
    __syncthreads();
    y[row * DMODEL + t] = dv * rsqrtf(stat[1] + 1e-5f) * g[t] + bb[t];
}

__global__ __launch_bounds__(128) void softmax_k(float* S, int cols, float scale, int causal, int M)
{
    long row = blockIdx.x;
    float* r = S + row * (long)cols;
    int pos = (int)(row % M);
    __shared__ float s[128];
    int t = threadIdx.x;
    float mx = -1e30f;
    for (int c = t; c < cols; c += 128) {
        float v = r[c] * scale;
        if (causal && c > pos) v = -1e9f;
        r[c] = v;
        mx = fmaxf(mx, v);
    }
    s[t] = mx; __syncthreads();
    for (int o = 64; o > 0; o >>= 1) { if (t < o) s[t] = fmaxf(s[t], s[t + o]); __syncthreads(); }
    mx = s[0]; __syncthreads();
    float sum = 0.f;
    for (int c = t; c < cols; c += 128) { float e = expf(r[c] - mx); r[c] = e; sum += e; }
    s[t] = sum; __syncthreads();
    for (int o = 64; o > 0; o >>= 1) { if (t < o) s[t] += s[t + o]; __syncthreads(); }
    float inv = 1.f / s[0];
    for (int c = t; c < cols; c += 128) r[c] *= inv;
}

__global__ void split_heads_k(const float* x, float* y, int L, long n)
{   // (B,L,H,Dh) -> (B*H, L, Dh)
    long i = (long)blockIdx.x * 256 + threadIdx.x; if (i >= n) return;
    int dd = (int)(i % DHEAD); long r = i / DHEAD;
    int h = (int)(r % NHEADS); r /= NHEADS;
    int l = (int)(r % L); int b = (int)(r / L);
    y[(((long)(b * NHEADS + h) * L + l) * DHEAD) + dd] = x[i];
}

__global__ void merge_heads_k(const float* x, float* y, int L, long n)
{   // (B*H, L, Dh) -> (B,L,H*Dh)
    long i = (long)blockIdx.x * 256 + threadIdx.x; if (i >= n) return;
    int dd = (int)(i % DHEAD); long r = i / DHEAD;
    int h = (int)(r % NHEADS); r /= NHEADS;
    int l = (int)(r % L); int b = (int)(r / L);
    y[i] = x[(((long)(b * NHEADS + h) * L + l) * DHEAD) + dd];
}

__global__ void gather_ch_k(const float* x, float* y, long n, int nch, int ch)
{
    long i = (long)blockIdx.x * 256 + threadIdx.x; if (i >= n) return;
    y[i] = x[i * nch + ch];
}

__global__ void stride_rows_k(const float* x, float* y, int Lin, int Lo, int start, int step, int d, long n)
{
    long i = (long)blockIdx.x * 256 + threadIdx.x; if (i >= n) return;
    int c = (int)(i % d); long r = i / d;
    int l = (int)(r % Lo); int b = (int)(r / Lo);
    y[i] = x[((long)b * Lin + start + l * step) * d + c];
}

__global__ void avgpool_k(const float* x, float* y, int Lin, int s, int Lo, int outL, int outOfs, int d, long n)
{
    long i = (long)blockIdx.x * 256 + threadIdx.x; if (i >= n) return;
    int c = (int)(i % d); long r = i / d;
    int l = (int)(r % Lo); int b = (int)(r / Lo);
    float acc = 0.f;
    for (int j = 0; j < s; ++j) acc += x[((long)b * Lin + l * s + j) * d + c];
    y[((long)b * outL + outOfs + l) * d + c] = acc / (float)s;
}

__global__ void upsample_k(const float* x, float* y, int L, int f, int d, long n)
{
    long i = (long)blockIdx.x * 256 + threadIdx.x; if (i >= n) return;
    int c = (int)(i % d); long r = i / d;
    int Lo = L * f;
    int lo = (int)(r % Lo); int b = (int)(r / Lo);
    float pos = ((float)lo + 0.5f) / (float)f - 0.5f;
    pos = fminf(fmaxf(pos, 0.f), (float)(L - 1));
    int l0 = (int)floorf(pos);
    int l1 = l0 + 1; if (l1 > L - 1) l1 = L - 1;
    float w = pos - (float)l0;
    const float* xb = x + (long)b * L * d;
    y[i] = xb[(long)l0 * d + c] * (1.f - w) + xb[(long)l1 * d + c] * w;
}

__global__ void mean2_k(const float* a, int La, int aRows, const float* b2, int Lb, int bRows,
                        float* mean, int d, long n)
{
    long i = (long)blockIdx.x * 256 + threadIdx.x; if (i >= n) return;
    int c = (int)(i % d); int b = (int)(i / d);
    float s = 0.f;
    for (int l = 0; l < La; ++l) s += a[((long)b * aRows + l) * d + c];
    for (int l = 0; l < Lb; ++l) s += b2[((long)b * bRows + l) * d + c];
    mean[i] = s / (float)(La + Lb);
}

__global__ void addmean_k(float* x, const float* mean, float sign, int L, int d, long n)
{
    long i = (long)blockIdx.x * 256 + threadIdx.x; if (i >= n) return;
    int c = (int)(i % d); int b = (int)(i / ((long)L * d));
    x[i] += sign * mean[(long)b * d + c];
}

__global__ void mark_emb_k(const float* mark, const float* temp, float* y, long n)
{
    long i = (long)blockIdx.x * 256 + threadIdx.x; if (i >= n) return;
    int c = (int)(i % DMODEL); long r = i / DMODEL;
    float s = 0.f;
    for (int j = 0; j < NMARK; ++j) s += mark[r * NMARK + j] * temp[(long)j * DMODEL + c];
    y[i] += s;
}

__global__ void posenc_add_k(float* y, int L, long n)
{
    long i = (long)blockIdx.x * 256 + threadIdx.x; if (i >= n) return;
    int c = (int)(i % DMODEL); int l = (int)((i / DMODEL) % L);
    int pair = c >> 1;
    float div = expf(-(float)(2 * pair) * logf(10000.f) / (float)DMODEL);
    float ang = (float)l * div;
    y[i] += (c & 1) ? cosf(ang) : sinf(ang);
}

__global__ __launch_bounds__(256) void chan_stats_k(const float* y, float* mv, int rows)
{
    int c = blockIdx.x; int t = threadIdx.x;
    __shared__ float s1[256]; __shared__ float s2[256];
    float a = 0.f, b = 0.f;
    for (int r = t; r < rows; r += 256) { float v = y[(long)r * DMODEL + c]; a += v; b += v * v; }
    s1[t] = a; s2[t] = b; __syncthreads();
    for (int o = 128; o > 0; o >>= 1) { if (t < o) { s1[t] += s1[t + o]; s2[t] += s2[t + o]; } __syncthreads(); }
    if (t == 0) { float m = s1[0] / rows; mv[c] = m; mv[DMODEL + c] = s2[0] / rows - m * m; }
}

__global__ void bn_elu_k(float* y, const float* mv, const float* g, const float* beta, long n)
{
    long i = (long)blockIdx.x * 256 + threadIdx.x; if (i >= n) return;
    int c = (int)(i % DMODEL);
    float v = (y[i] - mv[c]) * rsqrtf(mv[DMODEL + c] + 1e-5f) * g[c] + beta[c];
    y[i] = eluf(v);
}

__global__ void maxpool_k(const float* x, float* y, int L, int Lo, long n)
{
    long i = (long)blockIdx.x * 256 + threadIdx.x; if (i >= n) return;
    int c = (int)(i % DMODEL); long r = i / DMODEL;
    int lo = (int)(r % Lo); int b = (int)(r / Lo);
    int l0 = lo * 2 - 1;
    float m = -1e30f;
    for (int j = 0; j < 3; ++j) {
        int l = l0 + j;
        if (l >= 0 && l < L) m = fmaxf(m, x[((long)b * L + l) * DMODEL + c]);
    }
    y[i] = m;
}

__global__ void copy_cols_k(const float* src, float* dst, int dstLd, int colOfs, long n)
{
    long i = (long)blockIdx.x * 256 + threadIdx.x; if (i >= n) return;
    int c = (int)(i % DMODEL); long r = i / DMODEL;
    dst[r * dstLd + colOfs + c] = src[i];
}

__global__ void grouped_ffn_k(const float* o0, const float* o1, const float* o2,
                              const float* W1, const float* b1,
                              const float* W2, const float* b2,
                              float* allc, long n)
{
    long i = (long)blockIdx.x * 256 + threadIdx.x; if (i >= n) return;
    int g = (int)(i % DMODEL); long r = i / DMODEL;
    float x0 = o0[i], x1 = o1[i], x2 = o2[i];
    float a0 = 0.f, a1 = 0.f, a2 = 0.f;
    const float* w1 = W1 + (long)g * CONVDFF * 3;
    const float* w2 = W2 + (long)g * 3 * CONVDFF;
    const float* bb1 = b1 + (long)g * CONVDFF;
    for (int c = 0; c < CONVDFF; ++c) {
        float h = w1[c * 3 + 0] * x0 + w1[c * 3 + 1] * x1 + w1[c * 3 + 2] * x2 + bb1[c];
        h = geluf(h);
        a0 += w2[0 * CONVDFF + c] * h;
        a1 += w2[1 * CONVDFF + c] * h;
        a2 += w2[2 * CONVDFF + c] * h;
    }
    float* dst = allc + r * (3 * DMODEL) + g * 3;
    dst[0] = a0 + b2[g * 3 + 0];
    dst[1] = a1 + b2[g * 3 + 1];
    dst[2] = a2 + b2[g * 3 + 2];
}

__global__ void scatter_out_k(const float* c0, const float* c1, const float* c2,
                              float* out, int L, int Lp, long n)
{
    long i = (long)blockIdx.x * 256 + threadIdx.x; if (i >= n) return;
    int c = (int)(i % DMODEL); long r = i / DMODEL;
    int l = (int)(r % Lp); int b = (int)(r / Lp);
    long src = ((long)b * L + (L - Lp) + l) * DMODEL + c;
    out[i * 3 + 0] = c0[src];
    out[i * 3 + 1] = c1[src];
    out[i * 3 + 2] = c2[src];
}

// ---------------- host-side scaffolding ----------------
struct LinW  { const float* W; const float* b; };
struct LNW   { const float* b; const float* g; };
struct AttnW { LinW k, o, q, v; };
struct EncLayerW { AttnW attn; LinW ff1, ff2; LNW ln1, ln2; };
struct DecLayerW { AttnW cross; LinW ff1, ff2; LNW ln1, ln2, ln3; AttnW self_; };
struct ConvW { const float* W; const float* b; const float* beta; const float* g; };
struct EmbW  { const float* W; const float* temp; };
struct ModelW {
    DecLayerW dec0; LNW decLN; EmbW dec_emb;
    ConvW conv; EncLayerW enc[2]; LNW encLN; EmbW enc_emb;
    LinW ffn1, ffn2; LinW mlp[3]; LinW space;
};

// JAX pytree flatten order: dict keys sorted, lists in order.
static void load_params(void* const* d_in, ModelW& w)
{
    int c = 4;
    auto nx   = [&]() -> const float* { return (const float*)d_in[c++]; };
    auto lin  = [&](LinW& l) { l.W = nx(); l.b = nx(); };
    auto lnw  = [&](LNW& l) { l.b = nx(); l.g = nx(); };
    auto attn = [&](AttnW& a) { lin(a.k); lin(a.o); lin(a.q); lin(a.v); };
    attn(w.dec0.cross);
    lin(w.dec0.ff1); lin(w.dec0.ff2);
    lnw(w.dec0.ln1); lnw(w.dec0.ln2); lnw(w.dec0.ln3);
    attn(w.dec0.self_);
    lnw(w.decLN);
    w.dec_emb.W = nx(); w.dec_emb.temp = nx();
    w.conv.W = nx(); w.conv.b = nx(); w.conv.beta = nx(); w.conv.g = nx();
    for (int l = 0; l < 2; ++l) {
        attn(w.enc[l].attn);
        lin(w.enc[l].ff1); lin(w.enc[l].ff2);
        lnw(w.enc[l].ln1); lnw(w.enc[l].ln2);
    }
    lnw(w.encLN);
    w.enc_emb.W = nx(); w.enc_emb.temp = nx();
    lin(w.ffn1); lin(w.ffn2);
    lin(w.mlp[0]); lin(w.mlp[1]); lin(w.mlp[2]);
    lin(w.space);
}

struct Ws {
    char* base; size_t off;
    float* alloc(size_t nf) {
        float* p = (float*)(base + off);
        off = (off + nf * sizeof(float) + 255) & ~(size_t)255;
        return p;
    }
};

static inline dim3 g1(long n) { return dim3((unsigned)((n + 255) / 256)); }

static void gemm(hipStream_t st, const float* A, const float* B, const float* bias,
                 const float* R, float* C, int M, int N, int K, int batch,
                 long sA, long sB, long sC, long sR,
                 int lda, int ldb, int ldc, int bt, int act, int accum,
                 int rowShift = 0, int rowMod = 0)
{
    GemmP p;
    p.A = A; p.B = B; p.bias = bias; p.R = R; p.C = C;
    p.M = M; p.N = N; p.K = K;
    p.lda = lda; p.ldb = ldb; p.ldc = ldc;
    p.sA = sA; p.sB = sB; p.sC = sC; p.sR = sR;
    p.rowShift = rowShift; p.rowMod = rowMod;
    p.act = act; p.accum = accum;
    dim3 grid((N + BN - 1) / BN, (M + BM - 1) / BM, batch);
    if (rowMod > 0) {
        if (bt) gemm_f16_wmma<1, true ><<<grid, dim3(256), 0, st>>>(p);
        else    gemm_f16_wmma<0, true ><<<grid, dim3(256), 0, st>>>(p);
    } else {
        if (bt) gemm_f16_wmma<1, false><<<grid, dim3(256), 0, st>>>(p);
        else    gemm_f16_wmma<0, false><<<grid, dim3(256), 0, st>>>(p);
    }
}

static void layernorm(hipStream_t st, const float* x, LNW ln, float* y, int rows)
{
    layernorm_k<<<dim3((unsigned)rows), dim3(DMODEL), 0, st>>>(x, ln.g, ln.b, y);
}

static void attn_forward(Ws& ws, hipStream_t st, const float* xq, int Lq,
                         const float* xkv, int Ls, const AttnW& a, int causal,
                         const float* R, float* dst)
{
    size_t mk = ws.off;
    int Mq = BATCH * Lq, Ms = BATCH * Ls;
    float* Q = ws.alloc((size_t)Mq * DMODEL);
    float* K = ws.alloc((size_t)Ms * DMODEL);
    float* V = ws.alloc((size_t)Ms * DMODEL);
    gemm(st, xq,  a.q.W, a.q.b, nullptr, Q, Mq, DMODEL, DMODEL, 1, 0,0,0,0, DMODEL, DMODEL, DMODEL, 0, 0, 0);
    gemm(st, xkv, a.k.W, a.k.b, nullptr, K, Ms, DMODEL, DMODEL, 1, 0,0,0,0, DMODEL, DMODEL, DMODEL, 0, 0, 0);
    gemm(st, xkv, a.v.W, a.v.b, nullptr, V, Ms, DMODEL, DMODEL, 1, 0,0,0,0, DMODEL, DMODEL, DMODEL, 0, 0, 0);
    float* Qp = ws.alloc((size_t)Mq * DMODEL);
    float* Kp = ws.alloc((size_t)Ms * DMODEL);
    float* Vp = ws.alloc((size_t)Ms * DMODEL);
    long nq = (long)Mq * DMODEL, nk = (long)Ms * DMODEL;
    split_heads_k<<<g1(nq), 256, 0, st>>>(Q, Qp, Lq, nq);
    split_heads_k<<<g1(nk), 256, 0, st>>>(K, Kp, Ls, nk);
    split_heads_k<<<g1(nk), 256, 0, st>>>(V, Vp, Ls, nk);
    int nb = BATCH * NHEADS;
    float* S = ws.alloc((size_t)nb * Lq * Ls);
    // scores = Q . K^T : K is [Ls][Dh] == [N][K] layout -> bt=1 (no transpose staging)
    gemm(st, Qp, Kp, nullptr, nullptr, S, Lq, Ls, DHEAD, nb,
         (long)Lq * DHEAD, (long)Ls * DHEAD, (long)Lq * Ls, 0,
         DHEAD, DHEAD, Ls, 1, 0, 0);
    softmax_k<<<dim3((unsigned)(nb * Lq)), dim3(128), 0, st>>>(S, Ls, 0.17677669529663687f, causal, Lq);
    float* Op = ws.alloc((size_t)Mq * DMODEL);
    // out = P . V : V is [Ls][Dh] == [K][N] -> bt=0 (transpose-stage)
    gemm(st, S, Vp, nullptr, nullptr, Op, Lq, DHEAD, Ls, nb,
         (long)Lq * Ls, (long)Ls * DHEAD, (long)Lq * DHEAD, 0,
         Ls, DHEAD, DHEAD, 0, 0, 0);
    float* Om = ws.alloc((size_t)Mq * DMODEL);
    merge_heads_k<<<g1(nq), 256, 0, st>>>(Op, Om, Lq, nq);
    gemm(st, Om, a.o.W, a.o.b, R, dst, Mq, DMODEL, DMODEL, 1, 0,0,0,0, DMODEL, DMODEL, DMODEL, 0, 0, 0);
    ws.off = mk;
}

static void embed_fwd(hipStream_t st, const float* x, int L, const float* mark,
                      const EmbW& ep, float* dst)
{
    long sx = (long)L * DMODEL;
    // circular conv3 as 3 accumulating GEMMs with shifted rows
    gemm(st, x, ep.W + 0 * DMODEL * DMODEL, nullptr, nullptr, dst, L, DMODEL, DMODEL, BATCH,
         sx, 0, sx, 0, DMODEL, DMODEL, DMODEL, 0, 0, 0, -1, L);
    gemm(st, x, ep.W + 1 * DMODEL * DMODEL, nullptr, nullptr, dst, L, DMODEL, DMODEL, BATCH,
         sx, 0, sx, 0, DMODEL, DMODEL, DMODEL, 0, 0, 1, 0, L);
    gemm(st, x, ep.W + 2 * DMODEL * DMODEL, nullptr, nullptr, dst, L, DMODEL, DMODEL, BATCH,
         sx, 0, sx, 0, DMODEL, DMODEL, DMODEL, 0, 0, 1, 1, L);
    long n = (long)BATCH * L * DMODEL;
    mark_emb_k<<<g1(n), 256, 0, st>>>(mark, ep.temp, dst, n);
    posenc_add_k<<<g1(n), 256, 0, st>>>(dst, L, n);
}

static float* conv_fwd(Ws& ws, hipStream_t st, const float* x, int L, const ConvW& cv)
{
    int Lo = L / 2;
    float* out = ws.alloc((size_t)BATCH * Lo * DMODEL);
    size_t mk = ws.off;
    float* y = ws.alloc((size_t)BATCH * L * DMODEL);
    long sx = (long)L * DMODEL;
    gemm(st, x, cv.W + 0 * DMODEL * DMODEL, nullptr, nullptr, y, L, DMODEL, DMODEL, BATCH,
         sx, 0, sx, 0, DMODEL, DMODEL, DMODEL, 0, 0, 0, -1, L);
    gemm(st, x, cv.W + 1 * DMODEL * DMODEL, nullptr, nullptr, y, L, DMODEL, DMODEL, BATCH,
         sx, 0, sx, 0, DMODEL, DMODEL, DMODEL, 0, 0, 1, 0, L);
    gemm(st, x, cv.W + 2 * DMODEL * DMODEL, cv.b, nullptr, y, L, DMODEL, DMODEL, BATCH,
         sx, 0, sx, 0, DMODEL, DMODEL, DMODEL, 0, 0, 1, 1, L);
    float* mv = ws.alloc(2 * DMODEL);
    chan_stats_k<<<dim3(DMODEL), dim3(256), 0, st>>>(y, mv, BATCH * L);
    long n = (long)BATCH * L * DMODEL;
    bn_elu_k<<<g1(n), 256, 0, st>>>(y, mv, cv.g, cv.beta, n);
    long no = (long)BATCH * Lo * DMODEL;
    maxpool_k<<<g1(no), 256, 0, st>>>(y, out, L, Lo, no);
    ws.off = mk;
    return out;
}

static float* enc_layer_fwd(Ws& ws, hipStream_t st, const float* x, int L, const EncLayerW& el)
{
    int M = BATCH * L;
    float* t  = ws.alloc((size_t)M * DMODEL);
    attn_forward(ws, st, x, L, x, L, el.attn, 0, x, t);
    layernorm(st, t, el.ln1, t, M);
    float* t2 = ws.alloc((size_t)M * DMODEL);
    size_t mk = ws.off;
    float* y = ws.alloc((size_t)M * DFF);
    gemm(st, t, el.ff1.W, el.ff1.b, nullptr, y, M, DFF, DMODEL, 1, 0,0,0,0, DMODEL, DFF, DFF, 0, 1, 0);
    gemm(st, y, el.ff2.W, el.ff2.b, t, t2, M, DMODEL, DFF, 1, 0,0,0,0, DFF, DMODEL, DMODEL, 0, 0, 0);
    ws.off = mk;
    layernorm(st, t2, el.ln2, t2, M);
    return t2;
}

static float* encoder_fwd(Ws& ws, hipStream_t st, const float* x, int L, const ModelW& w, int* SeOut)
{
    float* x1 = enc_layer_fwd(ws, st, x, L, w.enc[0]);
    float* xc = conv_fwd(ws, st, x1, L, w.conv);
    int Lc = L / 2;
    float* x2 = enc_layer_fwd(ws, st, xc, Lc, w.enc[1]);
    layernorm(st, x2, w.encLN, x2, BATCH * Lc);
    *SeOut = Lc;
    return x2;
}

static float* decoder_fwd(Ws& ws, hipStream_t st, const float* x, int L,
                          const float* e, int Se, const DecLayerW& dl, LNW lnf)
{
    int M = BATCH * L;
    float* t1 = ws.alloc((size_t)M * DMODEL);
    attn_forward(ws, st, x, L, x, L, dl.self_, 1, x, t1);
    layernorm(st, t1, dl.ln1, t1, M);
    float* t2 = ws.alloc((size_t)M * DMODEL);
    attn_forward(ws, st, t1, L, e, Se, dl.cross, 0, t1, t2);
    layernorm(st, t2, dl.ln2, t2, M);
    float* t3 = ws.alloc((size_t)M * DMODEL);
    size_t mk = ws.off;
    float* y = ws.alloc((size_t)M * DFF);
    gemm(st, t2, dl.ff1.W, dl.ff1.b, nullptr, y, M, DFF, DMODEL, 1, 0,0,0,0, DMODEL, DFF, DFF, 0, 1, 0);
    gemm(st, y, dl.ff2.W, dl.ff2.b, t2, t3, M, DMODEL, DFF, 1, 0,0,0,0, DFF, DMODEL, DMODEL, 0, 0, 0);
    ws.off = mk;
    layernorm(st, t3, dl.ln3, t3, M);
    layernorm(st, t3, lnf, t3, M);
    return t3;
}

extern "C" void kernel_launch(void* const* d_in, const int* in_sizes, int n_in,
                              void* d_out, int out_size, void* d_ws, size_t ws_size,
                              hipStream_t stream)
{
    (void)in_sizes; (void)n_in; (void)out_size; (void)ws_size;
    const float* x_enc  = (const float*)d_in[0];
    const float* xm_enc = (const float*)d_in[1];
    const float* x_dec  = (const float*)d_in[2];
    const float* xm_dec = (const float*)d_in[3];
    ModelW w; load_params(d_in, w);
    Ws ws{ (char*)d_ws, 0 };
    hipStream_t st = stream;

    // persistent buffers
    float* xe[3]; float* xd[3]; float* coarse[3];
    size_t tok = (size_t)BATCH * SEQLEN * DMODEL;
    for (int i = 0; i < 3; ++i) { xe[i] = ws.alloc(tok); xd[i] = ws.alloc(tok); coarse[i] = ws.alloc(tok); }

    // ---- space projection per channel (512 -> 256) ----
    {
        size_t mk = ws.off;
        long nin = (long)BATCH * SEQLEN * SPACEDIM;
        float* ch = ws.alloc((size_t)nin);
        for (int i = 0; i < 3; ++i) {
            gather_ch_k<<<g1(nin), 256, 0, st>>>(x_enc, ch, nin, 3, i);
            gemm(st, ch, w.space.W, w.space.b, nullptr, xe[i], BATCH * SEQLEN, DMODEL, SPACEDIM, 1,
                 0,0,0,0, SPACEDIM, DMODEL, DMODEL, 0, 0, 0);
            gather_ch_k<<<g1(nin), 256, 0, st>>>(x_dec, ch, nin, 3, i);
            gemm(st, ch, w.space.W, w.space.b, nullptr, xd[i], BATCH * SEQLEN, DMODEL, SPACEDIM, 1,
                 0,0,0,0, SPACEDIM, DMODEL, DMODEL, 0, 0, 0);
        }
        ws.off = mk;
    }

    const int scales[3] = { 16, 4, 1 };
    int Lprev = 0;
    for (int si = 0; si < 3; ++si) {
        int sc = scales[si];
        int Lt = SEQLEN / sc;
        size_t smark = ws.off;
        long ntok = (long)BATCH * Lt * DMODEL;

        float* me = ws.alloc((size_t)BATCH * Lt * NMARK);
        float* md = ws.alloc((size_t)BATCH * Lt * NMARK);
        long nmk = (long)BATCH * Lt * NMARK;
        stride_rows_k<<<g1(nmk), 256, 0, st>>>(xm_enc, me, SEQLEN, Lt, sc / 2, sc, NMARK, nmk);
        stride_rows_k<<<g1(nmk), 256, 0, st>>>(xm_dec, md, SEQLEN, Lt, sc / 2, sc, NMARK, nmk);

        float* enc_o[3]; float* dec_o[3]; float* mn[3];
        for (int i = 0; i < 3; ++i) {
            enc_o[i] = ws.alloc((size_t)ntok);
            avgpool_k<<<g1(ntok), 256, 0, st>>>(xe[i], enc_o[i], SEQLEN, sc, Lt, Lt, 0, DMODEL, ntok);
            dec_o[i] = ws.alloc((size_t)ntok);
            if (si == 0) {
                avgpool_k<<<g1(ntok), 256, 0, st>>>(xd[i], dec_o[i], SEQLEN, sc, Lt, Lt, 0, DMODEL, ntok);
            } else {
                upsample_k<<<g1(ntok), 256, 0, st>>>(coarse[i], dec_o[i], Lprev, 4, DMODEL, ntok);
                int Ll = LABELLEN / sc;
                long nl2 = (long)BATCH * Ll * DMODEL;
                avgpool_k<<<g1(nl2), 256, 0, st>>>(xd[i], dec_o[i], SEQLEN, sc, Ll, Lt, 0, DMODEL, nl2);
            }
            int tail0 = LABELLEN / sc;
            mn[i] = ws.alloc((size_t)BATCH * DMODEL);
            long nbd = (long)BATCH * DMODEL;
            mean2_k<<<g1(nbd), 256, 0, st>>>(enc_o[i], Lt, Lt, dec_o[i] + (size_t)tail0 * DMODEL,
                                             Lt - tail0, Lt, mn[i], DMODEL, nbd);
            addmean_k<<<g1(ntok), 256, 0, st>>>(enc_o[i], mn[i], -1.f, Lt, DMODEL, ntok);
            addmean_k<<<g1(ntok), 256, 0, st>>>(dec_o[i], mn[i], -1.f, Lt, DMODEL, ntok);
        }

        float* outs[3];
        for (int i = 0; i < 3; ++i) {
            int full = (sc >= 4) || (i == 1);
            float* dd = ws.alloc((size_t)ntok);
            embed_fwd(st, dec_o[i], Lt, md, w.dec_emb, dd);
            if (full) {
                float* e0 = ws.alloc((size_t)ntok);
                embed_fwd(st, enc_o[i], Lt, me, w.enc_emb, e0);
                int Se = 0;
                float* e = encoder_fwd(ws, st, e0, Lt, w, &Se);
                outs[i] = decoder_fwd(ws, st, dd, Lt, e, Se, w.dec0, w.decLN);
            } else {
                outs[i] = dd;
            }
        }

        float* allc = ws.alloc((size_t)BATCH * Lt * 3 * DMODEL);
        if (sc >= 4) {
            for (int i = 0; i < 3; ++i)
                copy_cols_k<<<g1(ntok), 256, 0, st>>>(outs[i], allc, 3 * DMODEL, i * DMODEL, ntok);
        } else {
            grouped_ffn_k<<<g1(ntok), 256, 0, st>>>(outs[0], outs[1], outs[2],
                                                    w.ffn1.W, w.ffn1.b, w.ffn2.W, w.ffn2.b, allc, ntok);
        }
        for (int j = 0; j < 3; ++j) {
            gemm(st, allc, w.mlp[j].W, w.mlp[j].b, nullptr, coarse[j], BATCH * Lt, DMODEL, 3 * DMODEL, 1,
                 0,0,0,0, 3 * DMODEL, DMODEL, DMODEL, 0, 0, 0);
            addmean_k<<<g1(ntok), 256, 0, st>>>(coarse[j], mn[j], 1.f, Lt, DMODEL, ntok);
        }
        Lprev = Lt;
        ws.off = smark;
    }

    long nout = (long)BATCH * PREDLEN * DMODEL;
    scatter_out_k<<<g1(nout), 256, 0, st>>>(coarse[0], coarse[1], coarse[2],
                                            (float*)d_out, SEQLEN, PREDLEN, nout);
}
